// ModularACModel_52578989638115
// MI455X (gfx1250) — matrix operand order repaired and checked by reading, compiled, not run on hardware
//
#include <hip/hip_runtime.h>

#define BDIM 32768
#define FDIM 512
#define EDIM 8
#define HDIM 128
#define ADIM 256
#define TDIM 8
#define HCDIM 32

#define TILE 64                         // samples per block
#define MAX_TILES (BDIM / TILE + EDIM)  // 520 (padding: <=1 partial tile per expert)
#define NWAVES 4
#define NTHREADS 128

// ws layout (ints):
// [0..8)      counts
// [8..17)     offsets (9)
// [17..25)    cursor
// [32..552)   tileExpert
// [552..1072) tileBase
// [1072..)    sortedIdx (32768)
#define WS_COUNTS 0
#define WS_OFFS 8
#define WS_CURSOR 17
#define WS_TILE_E 32
#define WS_TILE_B 552
#define WS_SORTED 1072

typedef __attribute__((ext_vector_type(16))) __bf16 v16bf;
typedef __attribute__((ext_vector_type(8))) __bf16 v8bf;
typedef __attribute__((ext_vector_type(8))) float v8f;

// ---------------- bucketing kernels ----------------

__global__ void k_init(int* ws) {
  int t = threadIdx.x;
  if (t < EDIM) { ws[WS_COUNTS + t] = 0; ws[WS_CURSOR + t] = 0; }
}

__global__ void k_hist(const int* __restrict__ commands, int* __restrict__ ws) {
  int i = blockIdx.x * blockDim.x + threadIdx.x;
  if (i < BDIM) atomicAdd(&ws[WS_COUNTS + commands[i]], 1);
}

__global__ void k_plan(int* __restrict__ ws) {
  if (threadIdx.x == 0) {
    int off = 0;
    for (int e = 0; e < EDIM; ++e) { ws[WS_OFFS + e] = off; off += ws[WS_COUNTS + e]; }
    ws[WS_OFFS + EDIM] = off;
    int t = 0;
    for (int e = 0; e < EDIM; ++e) {
      int c = ws[WS_COUNTS + e];
      int nt = (c + TILE - 1) / TILE;
      for (int i = 0; i < nt; ++i) {
        ws[WS_TILE_E + t] = e;
        ws[WS_TILE_B + t] = ws[WS_OFFS + e] + i * TILE;
        ++t;
      }
    }
    for (; t < MAX_TILES; ++t) ws[WS_TILE_E + t] = -1;
  }
}

__global__ void k_scatter(const int* __restrict__ commands, int* __restrict__ ws) {
  int i = blockIdx.x * blockDim.x + threadIdx.x;
  if (i < BDIM) {
    int e = commands[i];
    int pos = atomicAdd(&ws[WS_CURSOR + e], 1);
    ws[WS_SORTED + ws[WS_OFFS + e] + pos] = i;
  }
}

// ---------------- fused GEMM kernel ----------------

__device__ __forceinline__ v16bf cat16(v8bf lo, v8bf hi) {
  return __builtin_shufflevector(lo, hi, 0, 1, 2, 3, 4, 5, 6, 7, 8, 9, 10, 11, 12, 13, 14, 15);
}

// A-fragment (16-bit, 16x32): lane<16 holds K {0..7,16..23}, lane>=16 holds K {8..15,24..31}.
// kbase = k0 + half*8.
__device__ __forceinline__ v16bf load_afrag_g(const float* __restrict__ row, int kbase) {
  const float4 a0 = *(const float4*)(row + kbase);
  const float4 a1 = *(const float4*)(row + kbase + 4);
  const float4 b0 = *(const float4*)(row + kbase + 16);
  const float4 b1 = *(const float4*)(row + kbase + 20);
  v16bf r;
  r[0] = (__bf16)a0.x;  r[1] = (__bf16)a0.y;  r[2] = (__bf16)a0.z;  r[3] = (__bf16)a0.w;
  r[4] = (__bf16)a1.x;  r[5] = (__bf16)a1.y;  r[6] = (__bf16)a1.z;  r[7] = (__bf16)a1.w;
  r[8] = (__bf16)b0.x;  r[9] = (__bf16)b0.y;  r[10] = (__bf16)b0.z; r[11] = (__bf16)b0.w;
  r[12] = (__bf16)b1.x; r[13] = (__bf16)b1.y; r[14] = (__bf16)b1.z; r[15] = (__bf16)b1.w;
  return r;
}

#define SLAB_STRIDE 40  // 32 K-values (bf16) + 8 pad per column; 80B, 16B-aligned

__global__ __launch_bounds__(NTHREADS, 1) void k_fused(
    const float* __restrict__ x, const int* __restrict__ tasks, const int* __restrict__ actions,
    const float* __restrict__ AW1, const float* __restrict__ Ab1,
    const float* __restrict__ AW2, const float* __restrict__ Ab2,
    const float* __restrict__ CW1, const float* __restrict__ Cb1,
    const float* __restrict__ CW2, const float* __restrict__ Cb2,
    const int* __restrict__ ws, float* __restrict__ out) {
  __shared__ __bf16 slab[ADIM * SLAB_STRIDE];     // 20 KB, B-fragment swizzled weight slab
  __shared__ __bf16 hbuf[NWAVES][16 * HDIM];      // 16 KB, h (bf16) per wave, [m][k]
  __shared__ float scratch[NWAVES][16 * ADIM];    // 64 KB, per-wave f32 scratch [m][n]

  const int e = ws[WS_TILE_E + blockIdx.x];
  if (e < 0) return;
  const int base = ws[WS_TILE_B + blockIdx.x];
  const int end = ws[WS_OFFS + e + 1];

  const int tid = threadIdx.x;
  const int lane = tid & 31;
  const int wave = tid >> 5;
  const int m = lane & 15;     // row within wave tile / column within n-tile
  const int half = lane >> 4;  // K-half selector

  const int rowpos = base + wave * 16 + m;
  const bool valid = rowpos < end;
  const int samp = ws[WS_SORTED + (valid ? rowpos : (end - 1))];
  const float* xrow = x + (size_t)samp * FDIM;

  // ================= actor layer 1: h = relu(x @ W1[e] + b1[e]) =================
  v8f acc1[8];
#pragma unroll
  for (int i = 0; i < 8; ++i) acc1[i] = {};
  const float* W1 = AW1 + (size_t)e * FDIM * HDIM;
  for (int kk = 0; kk < FDIM / 32; ++kk) {
    const int k0 = kk * 32;
    __syncthreads();
    {  // stage 32x128 slab, column-major bf16 (coalesced over n per k)
      const int n = tid;
#pragma unroll
      for (int i = 0; i < 32; ++i)
        slab[n * SLAB_STRIDE + i] = (__bf16)W1[(size_t)(k0 + i) * HDIM + n];
    }
    __syncthreads();
    const v16bf af = load_afrag_g(xrow, k0 + half * 8);
#pragma unroll
    for (int nt = 0; nt < 8; ++nt) {
      const __bf16* bp = &slab[(nt * 16 + m) * SLAB_STRIDE + half * 8];
      const v16bf bf = cat16(*(const v8bf*)bp, *(const v8bf*)(bp + 16));
      acc1[nt] = __builtin_amdgcn_wmma_f32_16x16x32_bf16(false, af, false, bf, (short)0,
                                                         acc1[nt], false, false);
    }
  }
  {  // bias + relu -> hbuf (bf16, [m][k])
    const float* b1 = Ab1 + e * HDIM;
#pragma unroll
    for (int nt = 0; nt < 8; ++nt) {
      const int n = nt * 16 + m;
      const float bv = b1[n];
#pragma unroll
      for (int r = 0; r < 8; ++r) {
        float hv = acc1[nt][r] + bv;
        hv = hv > 0.f ? hv : 0.f;
        hbuf[wave][(r + half * 8) * HDIM + n] = (__bf16)hv;
      }
    }
  }

  // ================= critic (dense over all T tasks): hc = relu(x @ Ccat + bcat) =================
  v8f acc2[16];
#pragma unroll
  for (int i = 0; i < 16; ++i) acc2[i] = {};
  for (int kk = 0; kk < FDIM / 32; ++kk) {
    const int k0 = kk * 32;
    __syncthreads();
#pragma unroll
    for (int c = 0; c < 2; ++c) {  // 256 columns; Ccat[k][n] = CW1[n/32][k][n%32]
      const int n = tid + c * NTHREADS;
      const int t = n >> 5, j = n & 31;
      const float* src = CW1 + ((size_t)t * FDIM + k0) * HCDIM + j;
#pragma unroll
      for (int i = 0; i < 32; ++i)
        slab[n * SLAB_STRIDE + i] = (__bf16)src[(size_t)i * HCDIM];
    }
    __syncthreads();
    const v16bf af = load_afrag_g(xrow, k0 + half * 8);
#pragma unroll
    for (int nt = 0; nt < 16; ++nt) {
      const __bf16* bp = &slab[(nt * 16 + m) * SLAB_STRIDE + half * 8];
      const v16bf bf = cat16(*(const v8bf*)bp, *(const v8bf*)(bp + 16));
      acc2[nt] = __builtin_amdgcn_wmma_f32_16x16x32_bf16(false, af, false, bf, (short)0,
                                                         acc2[nt], false, false);
    }
  }
  {  // bias + relu -> f32 scratch
#pragma unroll
    for (int nt = 0; nt < 16; ++nt) {
      const int n = nt * 16 + m;
      const float bv = Cb1[n];  // Cb1 flat [T*HC] == index n
#pragma unroll
      for (int r = 0; r < 8; ++r) {
        scratch[wave][(r + half * 8) * ADIM + n] = fmaxf(acc2[nt][r] + bv, 0.f);
      }
    }
  }
  // routed critic dot: 2 lanes per sample, 16 terms each
  const int t_id = tasks[samp];
  float vacc = 0.f;
  {
    const float* cw2 = CW2 + t_id * HCDIM;
    const float* hcrow = &scratch[wave][m * ADIM + t_id * HCDIM];
#pragma unroll
    for (int j = 0; j < 16; ++j) {
      const int jj = half * 16 + j;
      vacc += hcrow[jj] * cw2[jj];
    }
  }
  vacc += __shfl_xor(vacc, 16, 32);
  const float vcrit = vacc + Cb2[t_id];

  // ================= actor layer 2: scores = h @ W2[e] + b2[e] =================
  v8f acc3[16];
#pragma unroll
  for (int i = 0; i < 16; ++i) acc3[i] = {};
  const float* W2 = AW2 + (size_t)e * HDIM * ADIM;
  for (int kk = 0; kk < HDIM / 32; ++kk) {
    const int k0 = kk * 32;
    __syncthreads();
#pragma unroll
    for (int c = 0; c < 2; ++c) {
      const int n = tid + c * NTHREADS;
#pragma unroll
      for (int i = 0; i < 32; ++i)
        slab[n * SLAB_STRIDE + i] = (__bf16)W2[(size_t)(k0 + i) * ADIM + n];
    }
    __syncthreads();
    const int kb = k0 + half * 8;
    const __bf16* hp = &hbuf[wave][m * HDIM + kb];
    const v16bf af = cat16(*(const v8bf*)hp, *(const v8bf*)(hp + 16));
#pragma unroll
    for (int nt = 0; nt < 16; ++nt) {
      const __bf16* bp = &slab[(nt * 16 + m) * SLAB_STRIDE + half * 8];
      const v16bf bf = cat16(*(const v8bf*)bp, *(const v8bf*)(bp + 16));
      acc3[nt] = __builtin_amdgcn_wmma_f32_16x16x32_bf16(false, af, false, bf, (short)0,
                                                         acc3[nt], false, false);
    }
  }
  {  // bias -> f32 scratch (reuse; per-wave, wave-lockstep ordering is sufficient)
    const float* b2 = Ab2 + e * ADIM;
#pragma unroll
    for (int nt = 0; nt < 16; ++nt) {
      const int n = nt * 16 + m;
      const float bv = b2[n];
#pragma unroll
      for (int r = 0; r < 8; ++r)
        scratch[wave][(r + half * 8) * ADIM + n] = acc3[nt][r] + bv;
    }
  }

  // ================= log-softmax / entropy / routed outputs =================
  const float* row = &scratch[wave][m * ADIM];
  float mx = -3.4e38f;
#pragma unroll
  for (int i = 0; i < 128; i += 4) {
    const float4 q = *(const float4*)(row + half * 128 + i);
    mx = fmaxf(mx, fmaxf(fmaxf(q.x, q.y), fmaxf(q.z, q.w)));
  }
  mx = fmaxf(mx, __shfl_xor(mx, 16, 32));
  float se = 0.f, ses = 0.f;
#pragma unroll
  for (int i = 0; i < 128; i += 4) {
    const float4 q = *(const float4*)(row + half * 128 + i);
    const float d0 = q.x - mx, d1 = q.y - mx, d2 = q.z - mx, d3 = q.w - mx;
    const float e0 = __expf(d0), e1 = __expf(d1), e2 = __expf(d2), e3 = __expf(d3);
    se += e0 + e1 + e2 + e3;
    ses += e0 * d0 + e1 * d1 + e2 * d2 + e3 * d3;
  }
  se += __shfl_xor(se, 16, 32);
  ses += __shfl_xor(ses, 16, 32);
  const float lse = __logf(se);
  const float nege = ses / se - lse;
  const int a = actions[samp];
  const float lpc = (row[a] - mx) - lse;

  if (valid && half == 0) {
    out[(size_t)samp * 3 + 0] = lpc;
    out[(size_t)samp * 3 + 1] = vcrit;
    out[(size_t)samp * 3 + 2] = nege;
  }
}

extern "C" void kernel_launch(void* const* d_in, const int* in_sizes, int n_in,
                              void* d_out, int out_size, void* d_ws, size_t ws_size,
                              hipStream_t stream) {
  (void)in_sizes; (void)n_in; (void)out_size; (void)ws_size;
  const float* x = (const float*)d_in[0];
  const int* commands = (const int*)d_in[1];
  const int* tasks = (const int*)d_in[2];
  const int* actions = (const int*)d_in[3];
  const float* AW1 = (const float*)d_in[4];
  const float* Ab1 = (const float*)d_in[5];
  const float* AW2 = (const float*)d_in[6];
  const float* Ab2 = (const float*)d_in[7];
  const float* CW1 = (const float*)d_in[8];
  const float* Cb1 = (const float*)d_in[9];
  const float* CW2 = (const float*)d_in[10];
  const float* Cb2 = (const float*)d_in[11];
  int* ws = (int*)d_ws;
  float* out = (float*)d_out;

  hipLaunchKernelGGL(k_init, dim3(1), dim3(32), 0, stream, ws);
  hipLaunchKernelGGL(k_hist, dim3(BDIM / 256), dim3(256), 0, stream, commands, ws);
  hipLaunchKernelGGL(k_plan, dim3(1), dim3(32), 0, stream, ws);
  hipLaunchKernelGGL(k_scatter, dim3(BDIM / 256), dim3(256), 0, stream, commands, ws);
  hipLaunchKernelGGL(k_fused, dim3(MAX_TILES), dim3(NTHREADS), 0, stream,
                     x, tasks, actions, AW1, Ab1, AW2, Ab2, CW1, Cb1, CW2, Cb2, ws, out);
}
